// LocalAttentionBlock_15848429322815
// MI455X (gfx1250) — matrix operand order repaired and checked by reading, not compiled, not run on hardware
//
#include <hip/hip_runtime.h>
#include <hip/hip_bf16.h>

// ---------------------------------------------------------------------------
// MI455X (gfx1250) implementation: all matmuls via v_wmma_f32_16x16x32_f16
// (f16 inputs, fp32 accum), flash-style windowed attention, fused LN kernels.
// GEMMs use pre-transposed f16 weights so B-fragments are K-contiguous and
// load directly from global (L2-resident); only A is LDS-staged (8 KB).
// ---------------------------------------------------------------------------

typedef _Float16 half_t;
typedef __attribute__((ext_vector_type(16))) _Float16 v16h;
typedef __attribute__((ext_vector_type(8)))  _Float16 v8h;
typedef __attribute__((ext_vector_type(8)))  float    v8f;

#define DIM 512
#define HEADS 8
#define HDIM 64
#define FF 2048
#define WIN 128
#define NWIN 64
#define NB 8
#define NSEQ 8192
#define NROWS (NB * NSEQ)   // 65536

// ---------------------------------------------------------------------------
// WMMA helpers (CDNA5 16x16x32 f16 -> f32).  Fragment layouts per ISA 7.12.2:
//  A (16x32, 16-bit): lane l holds row m=l&15; half-group g=l>>4 holds
//      K = {8g..8g+7} in elems 0..7 and K = {16+8g..16+8g+7} in elems 8..15.
//  B (32x16, 16-bit): lane l holds col n=l&15; same K split as A.
//  C/D (16x16 f32): elem r, lane l -> M = r + 8*(l>>4), N = l&15.
// ---------------------------------------------------------------------------
__device__ inline v8f wmma16(v16h a, v16h b, v8f c) {
  return __builtin_amdgcn_wmma_f32_16x16x32_f16(false, a, false, b, (short)0, c,
                                                false, false);
}

// A rows must be >=16B aligned; lda in halves.
__device__ inline v16h load_a_frag(const half_t* __restrict__ A, int lda) {
  int lane = threadIdx.x & 31;
  int m  = lane & 15;
  int kh = (lane >> 4) * 8;
  v16h r;
  *(v8h*)&r       = *(const v8h*)(A + (size_t)m * lda + kh);
  *((v8h*)&r + 1) = *(const v8h*)(A + (size_t)m * lda + kh + 16);
  return r;
}

// Bt is K-contiguous per output column n (i.e. transposed weight layout).
__device__ inline v16h load_b_frag(const half_t* __restrict__ Bt, int ldb) {
  int lane = threadIdx.x & 31;
  int n  = lane & 15;
  int kb = (lane >> 4) * 8;
  v16h r;
  *(v8h*)&r       = *(const v8h*)(Bt + (size_t)n * ldb + kb);
  *((v8h*)&r + 1) = *(const v8h*)(Bt + (size_t)n * ldb + kb + 16);
  return r;
}

// ---------------------------------------------------------------------------
// fp32 [K][N] -> f16 [N][K] transposing conversion (LDS-tiled, one-time cost)
// ---------------------------------------------------------------------------
__global__ __launch_bounds__(256) void cvt_t_kernel(const float* __restrict__ in,
                                                    half_t* __restrict__ out,
                                                    int K, int N) {
  __shared__ half_t tile[32][33];
  int k0 = blockIdx.x * 32, n0 = blockIdx.y * 32;
  int tx = threadIdx.x & 31, ty = threadIdx.x >> 5;   // 32 x 8
#pragma unroll
  for (int i = 0; i < 4; ++i) {
    int k = k0 + ty + i * 8;
    tile[ty + i * 8][tx] = (half_t)in[(size_t)k * N + n0 + tx];
  }
  __syncthreads();
#pragma unroll
  for (int i = 0; i < 4; ++i) {
    int n = n0 + ty + i * 8;
    out[(size_t)n * K + k0 + tx] = tile[tx][ty + i * 8];
  }
}

// ---------------------------------------------------------------------------
// ss = silu(t_emb) @ time_w + time_b     (8 x 1024)
// ---------------------------------------------------------------------------
__global__ __launch_bounds__(256) void time_kernel(const float* __restrict__ t_emb,
                                                   const float* __restrict__ time_w,
                                                   const float* __restrict__ time_b,
                                                   float* __restrict__ ss) {
  __shared__ float s[DIM];
  int b = blockIdx.x, tid = threadIdx.x;
  for (int k = tid; k < DIM; k += 256) {
    float v = t_emb[b * DIM + k];
    s[k] = v / (1.f + __expf(-v));
  }
  __syncthreads();
  float acc[4] = {0.f, 0.f, 0.f, 0.f};
  for (int k = 0; k < DIM; ++k) {
    float sv = s[k];
    const float* wr = time_w + (size_t)k * (2 * DIM);
#pragma unroll
    for (int i = 0; i < 4; ++i) acc[i] += sv * wr[tid + 256 * i];
  }
#pragma unroll
  for (int i = 0; i < 4; ++i) {
    int j = tid + 256 * i;
    ss[b * (2 * DIM) + j] = acc[i] + time_b[j];
  }
}

// ---------------------------------------------------------------------------
// LayerNorm (+ optional adaLN modulation) -> f16.  One block (128 thr) / row.
// ---------------------------------------------------------------------------
__global__ __launch_bounds__(128) void ln_kernel(const float* __restrict__ xin,
                                                 const float* __restrict__ g,
                                                 const float* __restrict__ bb,
                                                 const float* __restrict__ ss,
                                                 half_t* __restrict__ out, int mod) {
  __shared__ float red[8];
  __shared__ float stats[2];
  int row = blockIdx.x;
  int b   = row >> 13;
  int tid = threadIdx.x;
  const float4 xv = *(const float4*)(xin + (size_t)row * DIM + tid * 4);
  float vals[4] = {xv.x, xv.y, xv.z, xv.w};
  float s = 0.f, sq = 0.f;
#pragma unroll
  for (int i = 0; i < 4; ++i) { s += vals[i]; sq += vals[i] * vals[i]; }
  for (int off = 16; off; off >>= 1) {
    s  += __shfl_down(s, off);
    sq += __shfl_down(sq, off);
  }
  int wid = tid >> 5;
  if ((tid & 31) == 0) { red[wid] = s; red[4 + wid] = sq; }
  __syncthreads();
  if (tid == 0) {
    float ts = red[0] + red[1] + red[2] + red[3];
    float tq = red[4] + red[5] + red[6] + red[7];
    float mu = ts * (1.f / DIM);
    float var = tq * (1.f / DIM) - mu * mu;
    stats[0] = mu;
    stats[1] = rsqrtf(var + 1e-5f);
  }
  __syncthreads();
  float mu = stats[0], rstd = stats[1];
#pragma unroll
  for (int i = 0; i < 4; ++i) {
    int c = tid * 4 + i;
    float xn = (vals[i] - mu) * rstd * g[c] + bb[c];
    if (mod) xn = xn * (1.f + ss[b * (2 * DIM) + c]) + ss[b * (2 * DIM) + DIM + c];
    out[(size_t)row * DIM + c] = (half_t)xn;
  }
}

// ---------------------------------------------------------------------------
// WMMA GEMM: C[M,N] = A[M,K](f16) * Bt[N,K](f16, pre-transposed) + bias.
// Block tile 128x128, 8 waves (2 M-groups x 4 N-groups), each wave 64x32:
//   per K-step: 4 A-frags (LDS), 2 B-frags (global/L2), 8 accum, 8 WMMAs.
// Epilogues:
//  MODE 0: split into head-major q/k/v f16 buffers
//  MODE 1: fp32 out = acc + bias + resid
//  MODE 2: f16 out = gelu(acc + bias)       (exact erf gelu)
//  MODE 3: fp32 out += acc + bias
// ---------------------------------------------------------------------------
template <int MODE>
__global__ __launch_bounds__(256) void gemm_kernel(
    const half_t* __restrict__ A, const half_t* __restrict__ Bt,
    const float* __restrict__ bias, int M, int N, int K,
    float* __restrict__ fout, const float* __restrict__ resid,
    half_t* __restrict__ hout,
    half_t* __restrict__ qo, half_t* __restrict__ ko, half_t* __restrict__ vo) {
  __shared__ half_t sA[128 * 32];   // 8 KB
  int bm = blockIdx.x * 128;
  int bn = blockIdx.y * 128;
  int tid = threadIdx.x;
  int wid = tid >> 5;
  int wm = (wid & 1) * 64;          // 0 / 64
  int wn = (wid >> 1) * 32;         // 0 / 32 / 64 / 96
  v8f acc[4][2];
#pragma unroll
  for (int ai = 0; ai < 4; ++ai)
#pragma unroll
    for (int bi = 0; bi < 2; ++bi) { v8f z = {}; acc[ai][bi] = z; }

  int sm = tid >> 1;                 // staging row 0..127
  int skk = (tid & 1) * 16;          // staging k 0 / 16
  const half_t* Arow = A + (size_t)(bm + sm) * K + skk;

  for (int k0 = 0; k0 < K; k0 += 32) {
    // stage A tile 128x32 (two b128 per thread)
    *(v8h*)&sA[sm * 32 + skk]     = *(const v8h*)(Arow + k0);
    *(v8h*)&sA[sm * 32 + skk + 8] = *(const v8h*)(Arow + k0 + 8);
    // prefetch next K-step of the streaming A matrix into caches
    if (k0 + 32 < K) __builtin_prefetch(Arow + k0 + 32, 0, 0);
    __syncthreads();
    v16h bf0 = load_b_frag(Bt + (size_t)(bn + wn) * K + k0, K);
    v16h bf1 = load_b_frag(Bt + (size_t)(bn + wn + 16) * K + k0, K);
#pragma unroll
    for (int ai = 0; ai < 4; ++ai) {
      v16h af = load_a_frag(sA + (wm + ai * 16) * 32, 32);
      acc[ai][0] = wmma16(af, bf0, acc[ai][0]);
      acc[ai][1] = wmma16(af, bf1, acc[ai][1]);
    }
    __syncthreads();
  }

  int lane = tid & 31;
  int lg = lane >> 4, ln = lane & 15;
#pragma unroll
  for (int ai = 0; ai < 4; ++ai) {
#pragma unroll
    for (int bi = 0; bi < 2; ++bi) {
#pragma unroll
      for (int r = 0; r < 8; ++r) {
        int row = bm + wm + ai * 16 + r + 8 * lg;
        int col = bn + wn + bi * 16 + ln;
        float val = acc[ai][bi][r] + bias[col];
        if constexpr (MODE == 0) {
          int which = col >> 9;
          int rem = col & 511;
          int head = rem >> 6;
          int d = rem & 63;
          size_t idx = (((size_t)(row >> 13) * HEADS + head) * NSEQ +
                        (size_t)(row & 8191)) * HDIM + d;
          half_t hv = (half_t)val;
          if (which == 0) qo[idx] = hv;
          else if (which == 1) ko[idx] = hv;
          else vo[idx] = hv;
        } else if constexpr (MODE == 1) {
          fout[(size_t)row * N + col] = val + resid[(size_t)row * N + col];
        } else if constexpr (MODE == 2) {
          float gl = 0.5f * val * (1.f + erff(val * 0.70710678118f));
          hout[(size_t)row * N + col] = (half_t)gl;
        } else {
          fout[(size_t)row * N + col] += val;
        }
      }
    }
  }
}

// ---------------------------------------------------------------------------
// Windowed (local) flash attention.  1 block per (b, head, window).
// 8 waves; wave owns 16 query rows.  Q/K fragments straight from global
// (head-major f16 layout is fragment-friendly); V staged transposed in LDS;
// P staged per-wave in LDS to reshape into A-fragments for P@V.
// ---------------------------------------------------------------------------
__global__ __launch_bounds__(256) void attn_kernel(const half_t* __restrict__ Q,
                                                   const half_t* __restrict__ Km,
                                                   const half_t* __restrict__ Vm,
                                                   half_t* __restrict__ aout) {
  __shared__ half_t sVt[HDIM * WIN];    // Vt[d][key]      16 KB
  __shared__ half_t sP[8][16 * WIN];    // per-wave P tile 32 KB
  int blk = blockIdx.x;
  int w  = blk & (NWIN - 1);
  int bh = blk >> 6;          // b*HEADS + h
  int h = bh & (HEADS - 1), b = bh >> 3;
  int tid = threadIdx.x, wid = tid >> 5, lane = tid & 31;
  int lg = lane >> 4, ln = lane & 15;

  const half_t* Qbase = Q + ((size_t)bh * NSEQ + (size_t)w * WIN + wid * 16) * HDIM;
  v16h qa0 = load_a_frag(Qbase, HDIM);        // K 0..31
  v16h qa1 = load_a_frag(Qbase + 32, HDIM);   // K 32..63

  float mrow[8], lrow[8];
  v8f o[4];
#pragma unroll
  for (int r = 0; r < 8; ++r) { mrow[r] = -1e30f; lrow[r] = 0.f; }
#pragma unroll
  for (int c = 0; c < 4; ++c) { v8f z = {}; o[c] = z; }

  for (int jw = w - 1; jw <= w + 1; ++jw) {
    if (jw < 0 || jw >= NWIN) continue;   // masked window -> zero weight
    __syncthreads();                      // protect sVt reuse
    {  // stage V transposed: sVt[d][key]
      int key = tid >> 1;
      int d0 = (tid & 1) * 32;
      const half_t* vp = Vm + ((size_t)bh * NSEQ + (size_t)jw * WIN + key) * HDIM + d0;
#pragma unroll
      for (int j = 0; j < 32; ++j) sVt[(d0 + j) * WIN + key] = vp[j];
    }
    // ---- scores S = Q @ K^T * d^-0.5 : eight 16x16 tiles per wave ----
    const half_t* Kbase = Km + ((size_t)bh * NSEQ + (size_t)jw * WIN) * HDIM;
    v8f st[8];
#pragma unroll
    for (int nt = 0; nt < 8; ++nt) {
      v16h b0 = load_b_frag(Kbase + (size_t)nt * 16 * HDIM, HDIM);
      v16h b1 = load_b_frag(Kbase + (size_t)nt * 16 * HDIM + 32, HDIM);
      v8f s = {};
      s = wmma16(qa0, b0, s);
      s = wmma16(qa1, b1, s);
#pragma unroll
      for (int r = 0; r < 8; ++r) s[r] *= 0.125f;   // 1/sqrt(64)
      st[nt] = s;
    }
    // ---- online softmax (row stats via 16-lane shfl_xor reductions) ----
    float corr[8];
#pragma unroll
    for (int r = 0; r < 8; ++r) {
      float cm = st[0][r];
#pragma unroll
      for (int nt = 1; nt < 8; ++nt) cm = fmaxf(cm, st[nt][r]);
      for (int msk = 1; msk < 16; msk <<= 1) cm = fmaxf(cm, __shfl_xor(cm, msk));
      float mn = fmaxf(mrow[r], cm);
      corr[r] = __expf(mrow[r] - mn);
      mrow[r] = mn;
    }
#pragma unroll
    for (int r = 0; r < 8; ++r) {
      float rs = 0.f;
      int prow = r + 8 * lg;
#pragma unroll
      for (int nt = 0; nt < 8; ++nt) {
        float p = __expf(st[nt][r] - mrow[r]);
        rs += p;
        sP[wid][prow * WIN + nt * 16 + ln] = (half_t)p;
      }
      for (int msk = 1; msk < 16; msk <<= 1) rs += __shfl_xor(rs, msk);
      lrow[r] = lrow[r] * corr[r] + rs;
#pragma unroll
      for (int c = 0; c < 4; ++c) o[c][r] *= corr[r];
    }
    __syncthreads();                      // sVt staged, sP written
    // ---- O += P @ V : K-dim 128 in four 32-steps ----
#pragma unroll
    for (int ks = 0; ks < 4; ++ks) {
      v16h pa = load_a_frag(&sP[wid][ks * 32], WIN);
#pragma unroll
      for (int c = 0; c < 4; ++c) {
        v16h vb = load_b_frag(sVt + (size_t)(c * 16) * WIN + ks * 32, WIN);
        o[c] = wmma16(pa, vb, o[c]);
      }
    }
  }
  // ---- normalize + write [b][n][h*64+d] f16 (proj-GEMM A layout) ----
  size_t nrow0 = (size_t)b * NSEQ + (size_t)w * WIN + wid * 16 + 8 * lg;
#pragma unroll
  for (int r = 0; r < 8; ++r) {
    size_t outbase = (nrow0 + r) * DIM + h * HDIM + ln;
    float inv = 1.f / lrow[r];
#pragma unroll
    for (int c = 0; c < 4; ++c) aout[outbase + c * 16] = (half_t)(o[c][r] * inv);
  }
}

// ---------------------------------------------------------------------------
// Host-side orchestration
// ---------------------------------------------------------------------------
extern "C" void kernel_launch(void* const* d_in, const int* in_sizes, int n_in,
                              void* d_out, int out_size, void* d_ws, size_t ws_size,
                              hipStream_t stream) {
  (void)in_sizes; (void)n_in; (void)out_size; (void)ws_size;
  const float* x      = (const float*)d_in[0];
  const float* t_emb  = (const float*)d_in[1];
  const float* ln1_g  = (const float*)d_in[2];
  const float* ln1_b  = (const float*)d_in[3];
  const float* qkv_w  = (const float*)d_in[4];
  const float* qkv_b  = (const float*)d_in[5];
  const float* proj_w = (const float*)d_in[6];
  const float* proj_b = (const float*)d_in[7];
  const float* ln2_g  = (const float*)d_in[8];
  const float* ln2_b  = (const float*)d_in[9];
  const float* mlp_w1 = (const float*)d_in[10];
  const float* mlp_b1 = (const float*)d_in[11];
  const float* mlp_w2 = (const float*)d_in[12];
  const float* mlp_b2 = (const float*)d_in[13];
  const float* time_w = (const float*)d_in[14];
  const float* time_b = (const float*)d_in[15];
  float* out = (float*)d_out;

  // ---- workspace carve (all section sizes multiples of 256B) ----
  char* p = (char*)d_ws;
  float* ss = (float*)p;            p += (size_t)NB * 2 * DIM * 4;          // 32 KB
  half_t* wqkv_t = (half_t*)p;      p += (size_t)DIM * 3 * DIM * 2;         // 1.5 MB  [1536][512]
  half_t* wproj_t = (half_t*)p;     p += (size_t)DIM * DIM * 2;             // 0.5 MB  [512][512]
  half_t* w1_t = (half_t*)p;        p += (size_t)DIM * FF * 2;              // 2 MB    [2048][512]
  half_t* w2_t = (half_t*)p;        p += (size_t)FF * DIM * 2;              // 2 MB    [512][2048]
  half_t* xn_h = (half_t*)p;        p += (size_t)NROWS * DIM * 2;           // 64 MB (reused as h2)
  half_t* q_h = (half_t*)p;         p += (size_t)NROWS * HDIM * HEADS * 2;  // 64 MB
  half_t* k_h = (half_t*)p;         p += (size_t)NROWS * HDIM * HEADS * 2;  // 64 MB
  half_t* v_h = (half_t*)p;         p += (size_t)NROWS * HDIM * HEADS * 2;  // 64 MB
  half_t* attn_h = (half_t*)p;      p += (size_t)NROWS * DIM * 2;           // 64 MB
  half_t* h2_h = xn_h;              // xn dead after QKV gemm
  half_t* h3_h = q_h;               // 256 MB alias over q/k/v/attn (dead after proj)

  // 1) weight conversion + transpose: f32 [K][N] -> f16 [N][K]
  cvt_t_kernel<<<dim3(DIM / 32, (3 * DIM) / 32), 256, 0, stream>>>(qkv_w, wqkv_t, DIM, 3 * DIM);
  cvt_t_kernel<<<dim3(DIM / 32, DIM / 32), 256, 0, stream>>>(proj_w, wproj_t, DIM, DIM);
  cvt_t_kernel<<<dim3(DIM / 32, FF / 32), 256, 0, stream>>>(mlp_w1, w1_t, DIM, FF);
  cvt_t_kernel<<<dim3(FF / 32, DIM / 32), 256, 0, stream>>>(mlp_w2, w2_t, FF, DIM);

  // 2) ss = silu(t_emb) @ time_w + time_b
  time_kernel<<<NB, 256, 0, stream>>>(t_emb, time_w, time_b, ss);

  // 3) x_norm = LN1(x) * (1+scale) + shift  -> f16
  ln_kernel<<<NROWS, 128, 0, stream>>>(x, ln1_g, ln1_b, ss, xn_h, 1);

  // 4) QKV gemm (65536 x 1536, K=512) -> head-major q/k/v f16
  gemm_kernel<0><<<dim3(NROWS / 128, (3 * DIM) / 128), 256, 0, stream>>>(
      xn_h, wqkv_t, qkv_b, NROWS, 3 * DIM, DIM, nullptr, nullptr, nullptr, q_h, k_h, v_h);

  // 5) local attention (4096 window-head blocks)
  attn_kernel<<<NB * HEADS * NWIN, 256, 0, stream>>>(q_h, k_h, v_h, attn_h);

  // 6) x1 = x + attn @ proj_w + proj_b  -> fp32 (d_out)
  gemm_kernel<1><<<dim3(NROWS / 128, DIM / 128), 256, 0, stream>>>(
      attn_h, wproj_t, proj_b, NROWS, DIM, DIM, out, x, nullptr, nullptr, nullptr, nullptr);

  // 7) h2 = LN2(x1) -> f16
  ln_kernel<<<NROWS, 128, 0, stream>>>(out, ln2_g, ln2_b, ss, h2_h, 0);

  // 8) h3 = gelu(h2 @ w1 + b1) -> f16 (65536 x 2048, K=512)
  gemm_kernel<2><<<dim3(NROWS / 128, FF / 128), 256, 0, stream>>>(
      h2_h, w1_t, mlp_b1, NROWS, FF, DIM, nullptr, nullptr, h3_h, nullptr, nullptr, nullptr);

  // 9) d_out += h3 @ w2 + b2 (65536 x 512, K=2048)
  gemm_kernel<3><<<dim3(NROWS / 128, DIM / 128), 256, 0, stream>>>(
      h3_h, w2_t, mlp_b2, NROWS, DIM, FF, out, nullptr, nullptr, nullptr, nullptr, nullptr);
}
